// CostVolume_1786706395662
// MI455X (gfx1250) — matrix operand order repaired and checked by reading, not compiled, not run on hardware
//
#include <hip/hip_runtime.h>

// CDNA5 / gfx1250 cost-volume via banded bf16 WMMA.
typedef __attribute__((ext_vector_type(16))) __bf16 v16bf;
typedef __attribute__((ext_vector_type(8)))  float  v8f;

constexpr int Bn = 16, Hn = 128, Wn = 128, Cn = 192, Rr = 4;
constexpr int TILES_W = Wn / 16;          // 8
constexpr int NDISP   = (2 * Rr + 1);     // 9
constexpr int OUT_C   = NDISP * NDISP;    // 81

// Convert 8 contiguous fp32 (16B-aligned) to 8 bf16 into dst[0..7].
__device__ __forceinline__ void cvt8(const float* __restrict__ p, __bf16* dst) {
  float4 q0 = ((const float4*)p)[0];
  float4 q1 = ((const float4*)p)[1];
  dst[0] = (__bf16)q0.x; dst[1] = (__bf16)q0.y;
  dst[2] = (__bf16)q0.z; dst[3] = (__bf16)q0.w;
  dst[4] = (__bf16)q1.x; dst[5] = (__bf16)q1.y;
  dst[6] = (__bf16)q1.z; dst[7] = (__bf16)q1.w;
}

__device__ __forceinline__ v16bf cvt16(const float* __restrict__ p) {
  __bf16 t[16];
  cvt8(p, t);
  cvt8(p + 8, t + 8);
  v16bf r;
#pragma unroll
  for (int j = 0; j < 16; ++j) r[j] = t[j];
  return r;
}

// K-loop over C=192 (six 16x16x32 bf16 WMMAs). MASKED=false is the fast path
// for fully in-bounds N-tiles (wave-uniform decision, no EXEC juggling).
template <bool MASKED>
__device__ __forceinline__ v8f kloop(const float* __restrict__ brow,
                                     const v16bf* __restrict__ afrag,
                                     bool ncol_ok) {
  v8f acc = {0.f, 0.f, 0.f, 0.f, 0.f, 0.f, 0.f, 0.f};
#pragma unroll
  for (int s = 0; s < 6; ++s) {
    v16bf bf;
    if (!MASKED || ncol_ok) {
      bf = cvt16(brow + 32 * s);
    } else {
#pragma unroll
      for (int j = 0; j < 16; ++j) bf[j] = (__bf16)0.0f;  // zero-pad columns
    }
    // EXEC is full here (any divergence above has reconverged).
    acc = __builtin_amdgcn_wmma_f32_16x16x32_bf16(
        false, afrag[s], false, bf, (short)0, acc, false, false);
  }
  return acc;
}

__global__ __launch_bounds__(64) void costvol_wmma_bf16(
    const float* __restrict__ prv, const float* __restrict__ nxt,
    float* __restrict__ out) {
  const int lane = threadIdx.x & 31;
  const int wave = threadIdx.x >> 5;      // 0 or 1: which N-tile

  // Decode (b, h, w-tile) from blockIdx.x
  const int tile = blockIdx.x;
  const int wt = tile & (TILES_W - 1);
  const int h  = (tile >> 3) & (Hn - 1);
  const int b  = tile >> 10;              // / (TILES_W * Hn)
  const int w0 = wt * 16;

  const int m  = lane & 15;               // A row (M) / B column (N) id
  const int kh = lane >> 4;               // K-subgroup selector (lane half)

  // This wave's N-tile column base and this lane's absolute nxt column.
  const int cb    = w0 - Rr + 16 * wave;  // [w0-4, w0+12) or [w0+12, w0+28)
  const int n_abs = cb + m;
  const bool ncol_ok = (n_abs >= 0) && (n_abs < Wn);
  // Wave-uniform, scalar-computable: is this whole N-tile in bounds?
  const bool tile_ok = (cb >= 0) && (cb + 15 < Wn);

  // ---- A fragments: 16 prv pixels x C, hoisted, reused for all 9 dy. ----
  // A in this code's contiguous k order (B matches): lane m holds K = 16*kh+[0,16).
  const float* aptr = prv + (((size_t)b * Hn + h) * Wn + (w0 + m)) * Cn;
  v16bf afrag[6];
#pragma unroll
  for (int s = 0; s < 6; ++s) {
    const int kb = 32 * s;
    __bf16 t[16];
    cvt8(aptr + kb +     16 * kh, t);
    cvt8(aptr + kb + 8 + 16 * kh, t + 8);
    v16bf a;
#pragma unroll
    for (int j = 0; j < 16; ++j) a[j] = t[j];
    afrag[s] = a;
  }

  const float scale = 1.0f / (float)Cn;

  for (int dyi = 0; dyi < NDISP; ++dyi) {
    const int hh = h + dyi - Rr;                  // uniform across the block
    float res[8];

    if (hh >= 0 && hh < Hn) {
      // Prefetch next dy row for this lane's column (global_prefetch_b8).
      if (dyi + 1 < NDISP) {
        const int hh2 = hh + 1;
        if (hh2 < Hn && ncol_ok)
          __builtin_prefetch(
              nxt + (((size_t)b * Hn + hh2) * Wn + n_abs) * Cn, 0, 0);
      }

      // B lane holds column n_abs, channels K = 16*kh + [0,16) per step.
      const float* brow =
          nxt + (((size_t)b * Hn + hh) * Wn + (ncol_ok ? n_abs : 0)) * Cn +
          16 * kh;

      v8f acc;
      if (tile_ok) {
        acc = kloop<false>(brow, afrag, true);    // common case: no masking
      } else {
        acc = kloop<true>(brow, afrag, ncol_ok);  // image edge: zeroed lanes
      }
#pragma unroll
      for (int v = 0; v < 8; ++v) res[v] = acc[v] * scale;
    } else {
#pragma unroll
      for (int v = 0; v < 8; ++v) res[v] = 0.0f;  // zero-padded dy row
    }

    // Band extraction. C/D layout: lane holds N = n_abs, rows M = v + 8*kh.
    #pragma unroll
    for (int v = 0; v < 8; ++v) {
      const int wm = w0 + v + 8 * kh;
      const int dx = n_abs - wm;
      if (dx >= -Rr && dx <= Rr) {
        out[(((size_t)b * Hn + h) * Wn + wm) * OUT_C + dyi * NDISP +
            (dx + Rr)] = res[v];
      }
    }
  }
}

extern "C" void kernel_launch(void* const* d_in, const int* in_sizes, int n_in,
                              void* d_out, int out_size, void* d_ws,
                              size_t ws_size, hipStream_t stream) {
  const float* prv = (const float*)d_in[0];
  const float* nxt = (const float*)d_in[1];
  // d_in[2] = search_range (4) — compiled in as Rr.
  float* out = (float*)d_out;

  const int grid = Bn * Hn * TILES_W;  // 16384 workgroups, 2 waves each
  costvol_wmma_bf16<<<grid, 64, 0, stream>>>(prv, nxt, out);
}
